// MVGRL_11854109737479
// MI455X (gfx1250) — compile-verified, matrix operand order checked by reading, not run on hardware
//
#include <hip/hip_runtime.h>
#include <hip/hip_bf16.h>
#include <stdint.h>

#define NN   50000
#define NE   800000
#define DIM  128
#define KAPP 8
#define BIGF ((size_t)NN * DIM)   // 6,400,000 floats

typedef __attribute__((ext_vector_type(16))) __bf16 bf16x16;
typedef __attribute__((ext_vector_type(8)))  float  f32x8;

union Frag { bf16x16 v; __bf16 b[16]; unsigned int u[8]; };
union BFBits { __bf16 h; unsigned short s; };

__device__ __forceinline__ unsigned short f32_to_bf16_bits(float f) {
    BFBits c; c.h = (__bf16)f;     // native RNE conversion on gfx1250
    return c.s;
}

// ---------------------------------------------------------------------------
// WMMA GEMM: out[M,128] = X[M,128] @ W[128,128] (+bias) (+PReLU) (+addin)
// block = 256 threads = 8 waves; each wave computes a 16x128 strip.
// W is staged into LDS as bf16, n-major (ldsW[n*128+k]) so B fragments are
// contiguous 16B LDS reads matching the wave32 WMMA operand layout.
// ---------------------------------------------------------------------------
template<int BIAS, int PRELU, int ADDIN>
__global__ __launch_bounds__(256) void gemm_xw(
    const float* __restrict__ X, const float* __restrict__ W,
    const float* __restrict__ bias, const float* __restrict__ addin,
    const float* __restrict__ alpha_p, float* __restrict__ out, int M)
{
    __shared__ unsigned short ldsW[DIM * DIM];  // 32 KB, bf16, n-major
    const int tid = threadIdx.x;
#pragma unroll
    for (int i = 0; i < 64; ++i) {
        int idx = i * 256 + tid;               // 0..16383
        int n = idx >> 7, k = idx & 127;
        ldsW[idx] = f32_to_bf16_bits(W[k * DIM + n]);
    }
    __syncthreads();

    const int wave = tid >> 5;
    const int lane = tid & 31;
    const int half = lane >> 4;                // K-octet selector
    const int l15  = lane & 15;
    const int rowBase = blockIdx.x * 128 + wave * 16;
    const int arow  = rowBase + l15;
    const int arowc = arow < M ? arow : (M - 1);
    const float* xrow = X + (size_t)arowc * DIM;
    __builtin_prefetch(xrow, 0, 0);            // global_prefetch_b8 (row = 1 cacheline stream)

    f32x8 acc[8] = {};
#pragma unroll
    for (int kt = 0; kt < 4; ++kt) {
        const int k0 = kt * 32 + half * 8;
        // A fragment: 16-bit A 16x32 layout — lane=M, VGPR0..3 hold K octet
        // {k0..k0+7}, VGPR4..7 hold {k0+16..k0+23}.
        Frag a;
        const float4* xa = (const float4*)(xrow + k0);
        const float4* xb = (const float4*)(xrow + k0 + 16);
        float4 p0 = xa[0], p1 = xa[1], p2 = xb[0], p3 = xb[1];
        float xs[16] = { p0.x, p0.y, p0.z, p0.w, p1.x, p1.y, p1.z, p1.w,
                         p2.x, p2.y, p2.z, p2.w, p3.x, p3.y, p3.z, p3.w };
#pragma unroll
        for (int j = 0; j < 16; ++j) a.b[j] = (__bf16)xs[j];
#pragma unroll
        for (int nt = 0; nt < 8; ++nt) {
            const int ncol = nt * 16 + l15;    // lane = N column
            const unsigned int* wp  = (const unsigned int*)&ldsW[ncol * DIM + k0];
            const unsigned int* wp2 = (const unsigned int*)&ldsW[ncol * DIM + k0 + 16];
            Frag b;
            b.u[0] = wp[0];  b.u[1] = wp[1];  b.u[2] = wp[2];  b.u[3] = wp[3];
            b.u[4] = wp2[0]; b.u[5] = wp2[1]; b.u[6] = wp2[2]; b.u[7] = wp2[3];
            acc[nt] = __builtin_amdgcn_wmma_f32_16x16x32_bf16(
                false, a.v, false, b.v, (short)0, acc[nt], false, false);
        }
    }

    float alpha = 0.0f;
    if (PRELU) alpha = *alpha_p;
    // D layout: VGPR v -> M = v + 8*half ; lane -> N = nt*16 + l15
#pragma unroll
    for (int nt = 0; nt < 8; ++nt) {
        const int col = nt * 16 + l15;
#pragma unroll
        for (int v = 0; v < 8; ++v) {
            const int row = rowBase + half * 8 + v;
            if (row < M) {
                float val = acc[nt][v];
                if (BIAS)  val += bias[col];
                if (PRELU) val = val >= 0.0f ? val : alpha * val;
                if (ADDIN) val += addin[(size_t)row * DIM + col];
                out[(size_t)row * DIM + col] = val;
            }
        }
    }
}

// ---------------------------------------------------------------------------
// Graph kernels
// ---------------------------------------------------------------------------
__global__ void degree_kernel(const int* __restrict__ src, const int* __restrict__ dst,
                              float* deg_out, float* deg_in, int E) {
    int e = blockIdx.x * 256 + threadIdx.x;
    if (e < E) {
        atomicAdd(&deg_out[src[e]], 1.0f);
        atomicAdd(&deg_in [dst[e]], 1.0f);
    }
}

__global__ void invsqrt_kernel(float* d, int n) {
    int i = blockIdx.x * 256 + threadIdx.x;
    if (i < n) d[i] = rsqrtf(fmaxf(d[i], 1.0f));
}

__global__ void scale_rows_kernel(const float* __restrict__ x,
                                  const float* __restrict__ norm,
                                  float* __restrict__ out) {
    size_t gid = (size_t)blockIdx.x * 256 + threadIdx.x;   // NN*DIM total
    out[gid] = x[gid] * norm[gid >> 7];
}

// 4 columns per thread: one b128 gather + 4 f32 atomics into L2
__global__ void scatter_add_kernel(const float* __restrict__ xn,
                                   const int* __restrict__ src,
                                   const int* __restrict__ dst,
                                   float* __restrict__ agg) {
    size_t gid = (size_t)blockIdx.x * 256 + threadIdx.x;   // NE*32 total
    int e  = (int)(gid >> 5);
    int c  = (int)(gid & 31) * 4;
    int s = src[e], d = dst[e];
    const float4 v = *(const float4*)&xn[(size_t)s * DIM + c];
    float* ap = &agg[(size_t)d * DIM + c];
    atomicAdd(ap + 0, v.x);
    atomicAdd(ap + 1, v.y);
    atomicAdd(ap + 2, v.z);
    atomicAdd(ap + 3, v.w);
}

// h = prelu(agg * norm_dst + b, alpha)
__global__ void gconv_fin_kernel(const float* __restrict__ agg,
                                 const float* __restrict__ norm_dst,
                                 const float* __restrict__ b,
                                 const float* __restrict__ alpha_p,
                                 float* __restrict__ out) {
    size_t gid = (size_t)blockIdx.x * 256 + threadIdx.x;
    int row = (int)(gid >> 7), col = (int)(gid & 127);
    float v = agg[gid] * norm_dst[row] + b[col];
    float a = *alpha_p;
    out[gid] = v >= 0.0f ? v : a * v;
}

// z = (1-alpha)*agg*norm_dst + alpha*feat
__global__ void appnp_fin_kernel(const float* __restrict__ agg,
                                 const float* __restrict__ norm_dst,
                                 const float* __restrict__ feat,
                                 float* __restrict__ out) {
    size_t gid = (size_t)blockIdx.x * 256 + threadIdx.x;
    int row = (int)(gid >> 7);
    out[gid] = 0.9f * agg[gid] * norm_dst[row] + 0.1f * feat[gid];
}

// column-sum pooling: out[128] += sum over a row chunk (out pre-zeroed)
__global__ void colsum_kernel(const float* __restrict__ x, float* __restrict__ out, int M) {
    int col = threadIdx.x;                 // blockDim = 128
    int r0 = blockIdx.x * 250;
    int r1 = r0 + 250; if (r1 > M) r1 = M;
    float s = 0.0f;
    for (int r = r0; r < r1; ++r) s += x[(size_t)r * DIM + col];
    atomicAdd(&out[col], s);
}

// tiny GEMV for the global MLP: out[128] = x[din] @ W[din,128] + b
template<int PRELU, int ADDIN>
__global__ void gemv_kernel(const float* __restrict__ x, const float* __restrict__ W,
                            const float* __restrict__ b, const float* __restrict__ addin,
                            const float* __restrict__ alpha_p,
                            float* __restrict__ out, int din) {
    __shared__ float xs[256];
    int t = threadIdx.x;                   // blockDim = 128
    for (int i = t; i < din; i += 128) xs[i] = x[i];
    __syncthreads();
    float s = b[t];
    for (int k = 0; k < din; ++k) s += xs[k] * W[k * DIM + t];
    if (PRELU) { float a = *alpha_p; s = s >= 0.0f ? s : a * s; }
    if (ADDIN) s += addin[t];
    out[t] = s;
}

// ---------------------------------------------------------------------------
// Host orchestration
// ---------------------------------------------------------------------------
extern "C" void kernel_launch(void* const* d_in, const int* in_sizes, int n_in,
                              void* d_out, int out_size, void* d_ws, size_t ws_size,
                              hipStream_t stream) {
    (void)in_sizes; (void)n_in; (void)out_size; (void)ws_size;

    const float* feat = (const float*)d_in[0];
    const int*   ei   = (const int*)d_in[1];
    const int*   esrc = ei;          // edge_index[0]
    const int*   edst = ei + NE;     // edge_index[1]

    // params flattened in setup_inputs() dict order
    const float* e1c0W = (const float*)d_in[2];
    const float* e1c0b = (const float*)d_in[3];
    const float* e1c1W = (const float*)d_in[4];
    const float* e1c1b = (const float*)d_in[5];
    const float* e1a   = (const float*)d_in[6];
    const float* e2lW  = (const float*)d_in[7];
    const float* e2lb  = (const float*)d_in[8];
    const float* e2la  = (const float*)d_in[9];
    const float* e2cW  = (const float*)d_in[10];
    const float* e2cb  = (const float*)d_in[11];
    const float* e2a   = (const float*)d_in[12];
    const float* lm1W  = (const float*)d_in[13];
    const float* lm1b  = (const float*)d_in[14];
    const float* lma1  = (const float*)d_in[15];
    const float* lm2W  = (const float*)d_in[16];
    const float* lm2b  = (const float*)d_in[17];
    const float* lma2  = (const float*)d_in[18];
    const float* lm3W  = (const float*)d_in[19];
    const float* lm3b  = (const float*)d_in[20];
    const float* lma3  = (const float*)d_in[21];
    const float* lmsW  = (const float*)d_in[22];
    const float* lmsb  = (const float*)d_in[23];
    const float* gm1W  = (const float*)d_in[24];   // [256,128]
    const float* gm1b  = (const float*)d_in[25];
    const float* gma1  = (const float*)d_in[26];
    const float* gm2W  = (const float*)d_in[27];
    const float* gm2b  = (const float*)d_in[28];
    const float* gma2  = (const float*)d_in[29];
    const float* gm3W  = (const float*)d_in[30];
    const float* gm3b  = (const float*)d_in[31];
    const float* gma3  = (const float*)d_in[32];
    const float* gmsW  = (const float*)d_in[33];   // [256,128]
    const float* gmsb  = (const float*)d_in[34];

    float* ws = (float*)d_ws;
    float* B0 = ws;
    float* B1 = ws + 1 * BIGF;
    float* B2 = ws + 2 * BIGF;
    float* B3 = ws + 3 * BIGF;
    float* nsrc = ws + 4 * BIGF;            // D_out^-1/2
    float* ndst = nsrc + NN;                // D_in^-1/2
    float* hg   = ndst + NN;                // 256
    float* hg2  = hg + 256;                 // 256
    float* g0   = hg2 + 256;                // 128
    float* g1   = g0 + 128;                 // 128

    float* out0 = (float*)d_out;            // mlp(local_v1) : 6,400,000
    float* out1 = out0 + BIGF;              // mlp(global_v1): 128
    float* out2 = out1 + 128;               // mlp(local_v2) : 6,400,000
    float* out3 = out2 + BIGF;              // mlp(global_v2): 128

    const dim3 blk256(256), blk128(128);
    const int gEdge   = NE / 256;                      // 3125
    const int gNode   = (NN + 255) / 256;              // 196
    const int gFeat   = (int)(BIGF / 256);             // 25000
    const int gScat   = (int)((size_t)NE * 32 / 256);  // 100000
    const int gGemm   = (NN + 127) / 128;              // 391
    const int gCol    = 200;

    // --- degrees -> norms ---
    hipMemsetAsync(nsrc, 0, 2 * NN * sizeof(float), stream);
    degree_kernel<<<gEdge, blk256, 0, stream>>>(esrc, edst, nsrc, ndst, NE);
    invsqrt_kernel<<<gNode, blk256, 0, stream>>>(nsrc, NN);
    invsqrt_kernel<<<gNode, blk256, 0, stream>>>(ndst, NN);

    // --- encoder1 layer 0: h = prelu(propagate(feat@W)+b) ---
    hipMemsetAsync(hg, 0, 512 * sizeof(float), stream);  // hg + hg2 zeroed
    gemm_xw<0,0,0><<<gGemm, blk256, 0, stream>>>(feat, e1c0W, nullptr, nullptr, nullptr, B0, NN);
    scale_rows_kernel<<<gFeat, blk256, 0, stream>>>(B0, nsrc, B1);
    hipMemsetAsync(B2, 0, BIGF * sizeof(float), stream);
    scatter_add_kernel<<<gScat, blk256, 0, stream>>>(B1, esrc, edst, B2);
    gconv_fin_kernel<<<gFeat, blk256, 0, stream>>>(B2, ndst, e1c0b, e1a, B0);   // h -> B0
    colsum_kernel<<<gCol, blk128, 0, stream>>>(B0, hg, NN);

    // --- encoder1 layer 1 -> local_v1 (B0) ---
    gemm_xw<0,0,0><<<gGemm, blk256, 0, stream>>>(B0, e1c1W, nullptr, nullptr, nullptr, B1, NN);
    scale_rows_kernel<<<gFeat, blk256, 0, stream>>>(B1, nsrc, B2);
    hipMemsetAsync(B3, 0, BIGF * sizeof(float), stream);
    scatter_add_kernel<<<gScat, blk256, 0, stream>>>(B2, esrc, edst, B3);
    gconv_fin_kernel<<<gFeat, blk256, 0, stream>>>(B3, ndst, e1c1b, e1a, B0);   // local_v1 -> B0
    colsum_kernel<<<gCol, blk128, 0, stream>>>(B0, hg + 128, NN);

    // --- mlp(local_v1) -> out0 ---
    gemm_xw<1,1,0><<<gGemm, blk256, 0, stream>>>(B0, lm1W, lm1b, nullptr, lma1, B1, NN);
    gemm_xw<1,1,0><<<gGemm, blk256, 0, stream>>>(B1, lm2W, lm2b, nullptr, lma2, B2, NN);
    gemm_xw<1,1,0><<<gGemm, blk256, 0, stream>>>(B2, lm3W, lm3b, nullptr, lma3, B1, NN);
    gemm_xw<1,0,1><<<gGemm, blk256, 0, stream>>>(B0, lmsW, lmsb, B1, nullptr, out0, NN);

    // --- mlp(global_v1 = hg[256]) -> out1 ---
    gemv_kernel<1,0><<<1, blk128, 0, stream>>>(hg, gm1W, gm1b, nullptr, gma1, g0, 256);
    gemv_kernel<1,0><<<1, blk128, 0, stream>>>(g0, gm2W, gm2b, nullptr, gma2, g1, 128);
    gemv_kernel<1,0><<<1, blk128, 0, stream>>>(g1, gm3W, gm3b, nullptr, gma3, g0, 128);
    gemv_kernel<0,1><<<1, blk128, 0, stream>>>(hg, gmsW, gmsb, g0, nullptr, out1, 256);

    // --- encoder2: APPNP(K=8, alpha=0.1), z kept in B1 ---
    const float* zsrc = feat;
    for (int k = 0; k < KAPP; ++k) {
        scale_rows_kernel<<<gFeat, blk256, 0, stream>>>(zsrc, nsrc, B2);
        hipMemsetAsync(B3, 0, BIGF * sizeof(float), stream);
        scatter_add_kernel<<<gScat, blk256, 0, stream>>>(B2, esrc, edst, B3);
        appnp_fin_kernel<<<gFeat, blk256, 0, stream>>>(B3, ndst, feat, B1);
        zsrc = B1;
    }
    // h = prelu(z @ Wlin + b)
    gemm_xw<1,1,0><<<gGemm, blk256, 0, stream>>>(B1, e2lW, e2lb, nullptr, e2la, B0, NN);
    colsum_kernel<<<gCol, blk128, 0, stream>>>(B0, hg2, NN);
    // gconv -> local_v2 (B0)
    gemm_xw<0,0,0><<<gGemm, blk256, 0, stream>>>(B0, e2cW, nullptr, nullptr, nullptr, B2, NN);
    scale_rows_kernel<<<gFeat, blk256, 0, stream>>>(B2, nsrc, B1);
    hipMemsetAsync(B3, 0, BIGF * sizeof(float), stream);
    scatter_add_kernel<<<gScat, blk256, 0, stream>>>(B1, esrc, edst, B3);
    gconv_fin_kernel<<<gFeat, blk256, 0, stream>>>(B3, ndst, e2cb, e2a, B0);
    colsum_kernel<<<gCol, blk128, 0, stream>>>(B0, hg2 + 128, NN);

    // --- mlp(local_v2) -> out2 ---
    gemm_xw<1,1,0><<<gGemm, blk256, 0, stream>>>(B0, lm1W, lm1b, nullptr, lma1, B1, NN);
    gemm_xw<1,1,0><<<gGemm, blk256, 0, stream>>>(B1, lm2W, lm2b, nullptr, lma2, B2, NN);
    gemm_xw<1,1,0><<<gGemm, blk256, 0, stream>>>(B2, lm3W, lm3b, nullptr, lma3, B1, NN);
    gemm_xw<1,0,1><<<gGemm, blk256, 0, stream>>>(B0, lmsW, lmsb, B1, nullptr, out2, NN);

    // --- mlp(global_v2 = hg2[256]) -> out3 ---
    gemv_kernel<1,0><<<1, blk128, 0, stream>>>(hg2, gm1W, gm1b, nullptr, gma1, g0, 256);
    gemv_kernel<1,0><<<1, blk128, 0, stream>>>(g0, gm2W, gm2b, nullptr, gma2, g1, 128);
    gemv_kernel<1,0><<<1, blk128, 0, stream>>>(g1, gm3W, gm3b, nullptr, gma3, g0, 128);
    gemv_kernel<0,1><<<1, blk128, 0, stream>>>(hg2, gmsW, gmsb, g0, nullptr, out3, 256);
}
